// BahdanauAttention_39213051413135
// MI455X (gfx1250) — compile-verified, matrix operand order checked by reading, not run on hardware
//
#include <hip/hip_runtime.h>
#include <hip/hip_bf16.h>

// Problem dims (fixed by reference)
#define NB    2048   // batch
#define NS    200    // seq len
#define NK    256    // Q_DIM == K_DIM
#define NV    256    // V_DIM
#define NA    128    // A_DIM

// Tiling: S padded to 13 tiles of 16 (208 rows); N = 128 = 8 waves x 16; K = 256 = 8 steps x 32
#define SROWS 208
#define KROW  264            // bf16 elements per padded keys row (256 + 8 pad -> conflict-free b128 reads)
#define EROW  129            // f32 elements per padded energy row

// LDS layout (bytes). E (200x129 f32 = 103200) overlaps sKeys region after a barrier.
#define OFF_SQ     109824    // 208*264*2
#define OFF_SQK    110848
#define OFF_SCORE  111360
#define OFF_ATTN   112384
#define OFF_RED    113408
#define LDS_BYTES  114432

#if __has_builtin(__builtin_amdgcn_tanhf)
#define FAST_TANH(x) __builtin_amdgcn_tanhf(x)
#else
#define FAST_TANH(x) tanhf(x)
#endif

typedef __attribute__((ext_vector_type(16))) __bf16 v16bf;
typedef __attribute__((ext_vector_type(8)))  float  v8f;

union ABu { uint4 q[2]; v16bf v; };

// f32 pair -> packed bf16 dword via the hardware packed converter (single VALU op,
// round-to-nearest-even). No clang builtin is declared for it on this toolchain,
// so emit the instruction directly.
__device__ __forceinline__ unsigned pk2(float lo, float hi) {
    unsigned r;
    asm("v_cvt_pk_bf16_f32 %0, %1, %2" : "=v"(r) : "v"(lo), "v"(hi));
    return r;
}

__global__ __launch_bounds__(256)
void bahdanau_fused_kernel(const float* __restrict__ query,
                           const float* __restrict__ keys,
                           const float* __restrict__ values,
                           const float* __restrict__ Wq_w,
                           const float* __restrict__ Wq_b,
                           const float* __restrict__ Wk_w,
                           const float* __restrict__ Wk_b,
                           const float* __restrict__ v_w,
                           float* __restrict__ out_ctx,    // [NB, NV]
                           float* __restrict__ out_attn)   // [NB, NS]
{
    extern __shared__ char smem[];
    __bf16* sKeys  = (__bf16*)smem;
    float*  sQ     = (float*)(smem + OFF_SQ);
    float*  sQK    = (float*)(smem + OFF_SQK);
    float*  sScore = (float*)(smem + OFF_SCORE);
    float*  sAttn  = (float*)(smem + OFF_ATTN);
    float*  sRed   = (float*)(smem + OFF_RED);

    const int b = blockIdx.x;
    const int t = threadIdx.x;

    // ---------- Phase 1: load query row; keys fp32 -> bf16 into LDS ----------
    sQ[t] = query[(size_t)b * NK + t];

    {
        const float* kb = keys + (size_t)b * NS * NK;
        // 200*256 = 51200 floats = 12800 float4; 50 per thread
        #pragma unroll 2
        for (int i = 0; i < 50; ++i) {
            int id  = t + 256 * i;          // float4 index
            int row = id >> 6;              // /64 float4 per row
            int col = (id & 63) << 2;       // element column
            float4 f = *(const float4*)(kb + row * NK + col);
            uint2 p  = make_uint2(pk2(f.x, f.y), pk2(f.z, f.w));
            *(uint2*)(sKeys + row * KROW + col) = p;
        }
        // zero-pad rows 200..207 (cols 0..255): thread t clears one uint4 (8 bf16)
        int zrow = 200 + (t >> 5);
        int zcol = (t & 31) << 3;
        *(uint4*)(sKeys + zrow * KROW + zcol) = make_uint4(0u, 0u, 0u, 0u);
    }
    __syncthreads();

    // ---------- Phase 2: qproj[a] = dot(query[b], Wq_w[a]) + Wq_b[a] + Wk_b[a] ----------
    if (t < NA) {
        const float* wq = Wq_w + t * NK;
        float acc = Wq_b[t] + Wk_b[t];
        #pragma unroll 4
        for (int k = 0; k < NK; k += 4) {
            float4 w = *(const float4*)(wq + k);
            acc += sQ[k] * w.x + sQ[k + 1] * w.y + sQ[k + 2] * w.z + sQ[k + 3] * w.w;
        }
        sQK[t] = acc;
    }
    // (no barrier needed: sQK first read after the post-GEMM barrier)

    // ---------- Phase 3: WMMA GEMM  kproj[s,a] = sum_k keys_bf16[s,k] * Wk_bf16[a,k] ----------
    const int wave = t >> 5;
    const int lane = t & 31;
    const int ln   = lane & 15;
    const int lh   = lane >> 4;

    const float* wkrow = Wk_w + (size_t)(wave * 16 + ln) * NK;   // B column = Wk row (a)

    v8f acc[13];
    #pragma unroll
    for (int m = 0; m < 13; ++m) acc[m] = (v8f)(0.0f);

    for (int kc = 0; kc < NK; kc += 32) {
        const int koff = kc + lh * 8;   // half-lane K split per ISA 16-bit fragment layout
        // B fragment: 16 fp32 from global Wk (L2-resident), pack to bf16
        float4 f0 = *(const float4*)(wkrow + koff);
        float4 f1 = *(const float4*)(wkrow + koff + 4);
        float4 f2 = *(const float4*)(wkrow + koff + 16);
        float4 f3 = *(const float4*)(wkrow + koff + 20);
        ABu bu;
        bu.q[0] = make_uint4(pk2(f0.x, f0.y), pk2(f0.z, f0.w), pk2(f1.x, f1.y), pk2(f1.z, f1.w));
        bu.q[1] = make_uint4(pk2(f2.x, f2.y), pk2(f2.z, f2.w), pk2(f3.x, f3.y), pk2(f3.z, f3.w));

        #pragma unroll
        for (int m = 0; m < 13; ++m) {
            const __bf16* ar = sKeys + (m * 16 + ln) * KROW + koff;
            ABu au;
            au.q[0] = *(const uint4*)(ar);        // K = koff..koff+7
            au.q[1] = *(const uint4*)(ar + 16);   // K = koff+16..koff+23
            acc[m] = __builtin_amdgcn_wmma_f32_16x16x32_bf16(
                false, au.v, false, bu.v, (short)0, acc[m], false, false);
        }
    }

    __syncthreads();   // all waves done reading sKeys; sQK ready

    // ---------- Phase 4: energy = tanh(kproj + qk); E[s][a] = energy * v_w[a] ----------
    {
        float* E = (float*)smem;                 // overlaps sKeys
        const int a  = wave * 16 + ln;
        const float qk = sQK[a];
        const float vw = v_w[a];
        #pragma unroll
        for (int m = 0; m < 13; ++m) {
            #pragma unroll
            for (int r = 0; r < 8; ++r) {
                const int s = m * 16 + r + lh * 8;
                if (s < NS) {
                    E[s * EROW + a] = FAST_TANH(acc[m][r] + qk) * vw;
                }
            }
        }
    }
    __syncthreads();

    // ---------- Phase 5: scores[s] = sum_a E[s][a]; softmax over s ----------
    {
        const float* E = (const float*)smem;
        float sc = -3.4e38f;
        if (t < NS) {
            float sum = 0.f;
            const float* row = E + t * EROW;
            #pragma unroll 4
            for (int i = 0; i < NA; ++i) sum += row[i];
            sc = sum;
        }
        sScore[t] = sc;
        sRed[t]   = sc;
    }
    for (int off = 128; off > 0; off >>= 1) {
        __syncthreads();
        if (t < off) sRed[t] = fmaxf(sRed[t], sRed[t + off]);
    }
    __syncthreads();
    const float mx = sRed[0];
    __syncthreads();
    {
        float e = 0.f;
        if (t < NS) e = __expf(sScore[t] - mx);
        sAttn[t] = e;
        sRed[t]  = e;
    }
    for (int off = 128; off > 0; off >>= 1) {
        __syncthreads();
        if (t < off) sRed[t] += sRed[t + off];
    }
    __syncthreads();
    const float inv = 1.0f / sRed[0];
    __syncthreads();
    if (t < NS) {
        const float w = sAttn[t] * inv;
        sAttn[t] = w;
        out_attn[(size_t)b * NS + t] = w;
    }
    __syncthreads();

    // ---------- Phase 6: context[v] = sum_s attn[s] * values[b,s,v] ----------
    {
        const float* vbase = values + (size_t)b * NS * NV + t;   // coalesced over t
        float c0 = 0.f, c1 = 0.f, c2 = 0.f, c3 = 0.f;
        #pragma unroll 2
        for (int s = 0; s < NS; s += 4) {
            c0 += sAttn[s]     * vbase[(s)     * NV];
            c1 += sAttn[s + 1] * vbase[(s + 1) * NV];
            c2 += sAttn[s + 2] * vbase[(s + 2) * NV];
            c3 += sAttn[s + 3] * vbase[(s + 3) * NV];
        }
        out_ctx[(size_t)b * NV + t] = (c0 + c1) + (c2 + c3);
    }
}

extern "C" void kernel_launch(void* const* d_in, const int* in_sizes, int n_in,
                              void* d_out, int out_size, void* d_ws, size_t ws_size,
                              hipStream_t stream) {
    (void)in_sizes; (void)n_in; (void)d_ws; (void)ws_size; (void)out_size;
    const float* query = (const float*)d_in[0];
    const float* keys  = (const float*)d_in[1];
    const float* values= (const float*)d_in[2];
    const float* Wq_w  = (const float*)d_in[3];
    const float* Wq_b  = (const float*)d_in[4];
    const float* Wk_w  = (const float*)d_in[5];
    const float* Wk_b  = (const float*)d_in[6];
    const float* v_w   = (const float*)d_in[7];

    float* out_ctx  = (float*)d_out;                       // [NB, NV]
    float* out_attn = (float*)d_out + (size_t)NB * NV;     // [NB, NS]

    bahdanau_fused_kernel<<<NB, 256, LDS_BYTES, stream>>>(
        query, keys, values, Wq_w, Wq_b, Wk_w, Wk_b, v_w, out_ctx, out_attn);
}
